// KalmanFilter_35888746726128
// MI455X (gfx1250) — matrix operand order repaired and checked by reading, not compiled, not run on hardware
//
#include <hip/hip_runtime.h>
#include <hip/hip_bf16.h>

typedef __attribute__((ext_vector_type(16))) _Float16 v16h;
typedef __attribute__((ext_vector_type(8)))  _Float16 v8h;
typedef __attribute__((ext_vector_type(4)))  _Float16 v4h;
typedef __attribute__((ext_vector_type(8)))  float    v8f;
typedef __attribute__((ext_vector_type(4)))  float    v4f;

#define N_FRAMES 8
#define DIMC     256
#define N_HEADS  4
#define HDIM     64
#define TOK      1024     // H*W tokens per frame
#define NROWS    8192     // N_FRAMES * TOK
#define FFH      2048     // 2*FF
#define FFO      1024     // FF

// ---------------------------------------------------------------------------
// WMMA helpers (CDNA5 wave32, v_wmma_f32_16x16x32_f16)
// ---------------------------------------------------------------------------
__device__ __forceinline__ v8f wmma_f16(v16h a, v16h b, v8f c) {
  return __builtin_amdgcn_wmma_f32_16x16x32_f16(
      /*neg_a=*/false, a, /*neg_b=*/false, b,
      /*c_mod=*/(short)0, c, /*reuse_a=*/false, /*reuse_b=*/false);
}

__device__ __forceinline__ v16h mk16(v8h lo, v8h hi) {
  v16h r;
#pragma unroll
  for (int i = 0; i < 8; ++i) { r[i] = lo[i]; r[i + 8] = hi[i]; }
  return r;
}

__device__ __forceinline__ v4h cvt4(v4f x) {
  v4h r;
#pragma unroll
  for (int j = 0; j < 4; ++j) r[j] = (_Float16)x[j];
  return r;
}

__device__ __forceinline__ float sigf(float x) { return 1.f / (1.f + __expf(-x)); }
__device__ __forceinline__ float geluf(float x) {
  return 0.5f * x * (1.f + tanhf(0.7978845608f * (x + 0.044715f * x * x * x)));
}

// Low 32 bits of an LDS-aperture generic pointer = LDS byte offset.
__device__ __forceinline__ unsigned lds_addr32(const void* p) {
  return (unsigned)(unsigned long long)p;
}

// ---------------------------------------------------------------------------
// Generic GEMM: C[M,N] = A[M,K] @ B (+bias, +residual). f32 in/out, f16 WMMA.
// Block 256 = 8 waves; macro-tile 128(M) x 64(N); per-wave 32x32 (4 WMMAs).
// K stepped by 32 through LDS; float4 global loads register-pipelined against
// the WMMA compute of the previous K-step. TRANSB is compile-time so the
// staging stores are branch-free.
// Requires M%128==0, N%64==0, K%32==0 (true for all uses here).
// ---------------------------------------------------------------------------
template <int TRANSB>
__global__ __launch_bounds__(256) void gemm_wmma(
    const float* __restrict__ A, const float* __restrict__ B,
    const float* __restrict__ bias, const float* __restrict__ res,
    float* __restrict__ C, int M, int N, int K) {
  __shared__ __align__(16) _Float16 sA[128 * 40];
  __shared__ __align__(16) _Float16 sB[64 * 40];
  const int t = threadIdx.x;
  const int lane = t & 31, wave = t >> 5;
  const int wm = wave & 3, wn = wave >> 2;       // 4 x 2 wave grid
  const int lh = lane & 15, hi = lane >> 4;
  const int mBase = blockIdx.y * 128;
  const int nBase = blockIdx.x * 64;
  const int r0 = wm * 32, c0 = wn * 32;

  v8f acc00 = {}, acc01 = {}, acc10 = {}, acc11 = {};
  v4f aR[4], bR[2];

  // ---- preload K-tile 0 into registers (float4 / global_load_b128) ----
#pragma unroll
  for (int i = 0; i < 4; ++i) {
    int q = t + i * 256;                 // 1024 quads = 128 rows x 8 quads
    int r = q >> 3, cq = (q & 7) * 4;
    aR[i] = *(const v4f*)&A[(size_t)(mBase + r) * K + cq];
  }
#pragma unroll
  for (int i = 0; i < 2; ++i) {
    int q = t + i * 256;                 // 512 quads
    if (TRANSB) {
      int n = q >> 3, ck = (q & 7) * 4;
      bR[i] = *(const v4f*)&B[(size_t)(nBase + n) * K + ck];
    } else {
      int kk = q >> 4, nq = (q & 15) * 4;
      bR[i] = *(const v4f*)&B[(size_t)kk * N + nBase + nq];
    }
  }

  for (int k0 = 0; k0 < K; k0 += 32) {
    // ---- commit staged registers to LDS (packed f16) ----
#pragma unroll
    for (int i = 0; i < 4; ++i) {
      int q = t + i * 256;
      int r = q >> 3, cq = (q & 7) * 4;
      *(v4h*)&sA[r * 40 + cq] = cvt4(aR[i]);
    }
#pragma unroll
    for (int i = 0; i < 2; ++i) {
      int q = t + i * 256;
      if (TRANSB) {
        int n = q >> 3, ck = (q & 7) * 4;
        *(v4h*)&sB[n * 40 + ck] = cvt4(bR[i]);
      } else {
        int kk = q >> 4, nq = (q & 15) * 4;
#pragma unroll
        for (int j = 0; j < 4; ++j) sB[(nq + j) * 40 + kk] = (_Float16)bR[i][j];
      }
    }
    __syncthreads();

    // ---- issue next K-tile's global loads while WMMAs run ----
    const int kn = k0 + 32;
    if (kn < K) {
#pragma unroll
      for (int i = 0; i < 4; ++i) {
        int q = t + i * 256;
        int r = q >> 3, cq = (q & 7) * 4;
        aR[i] = *(const v4f*)&A[(size_t)(mBase + r) * K + kn + cq];
      }
#pragma unroll
      for (int i = 0; i < 2; ++i) {
        int q = t + i * 256;
        if (TRANSB) {
          int n = q >> 3, ck = (q & 7) * 4;
          bR[i] = *(const v4f*)&B[(size_t)(nBase + n) * K + kn + ck];
        } else {
          int kk = q >> 4, nq = (q & 15) * 4;
          bR[i] = *(const v4f*)&B[(size_t)(kn + kk) * N + nBase + nq];
        }
      }
    }

    // ---- wave compute: 2 A-frags x 2 B-frags -> 4 WMMAs ----
    v8h a0lo = *(const v8h*)&sA[(r0 + lh) * 40 + hi * 16];
    v8h a0hi = *(const v8h*)&sA[(r0 + lh) * 40 + hi * 16 + 8];
    v8h a1lo = *(const v8h*)&sA[(r0 + 16 + lh) * 40 + hi * 16];
    v8h a1hi = *(const v8h*)&sA[(r0 + 16 + lh) * 40 + hi * 16 + 8];
    v8h b0lo = *(const v8h*)&sB[(c0 + lh) * 40 + hi * 16];
    v8h b0hi = *(const v8h*)&sB[(c0 + lh) * 40 + hi * 16 + 8];
    v8h b1lo = *(const v8h*)&sB[(c0 + 16 + lh) * 40 + hi * 16];
    v8h b1hi = *(const v8h*)&sB[(c0 + 16 + lh) * 40 + hi * 16 + 8];
    v16h A0 = mk16(a0lo, a0hi), A1 = mk16(a1lo, a1hi);
    v16h B0 = mk16(b0lo, b0hi), B1 = mk16(b1lo, b1hi);
    acc00 = wmma_f16(A0, B0, acc00);
    acc01 = wmma_f16(A0, B1, acc01);
    acc10 = wmma_f16(A1, B0, acc10);
    acc11 = wmma_f16(A1, B1, acc11);
    __syncthreads();
  }

  // ---- epilogue: bias + optional residual ----
  const int colA = nBase + c0 + lh;
  const int colB = nBase + c0 + 16 + lh;
  const float bvA = bias ? bias[colA] : 0.f;
  const float bvB = bias ? bias[colB] : 0.f;
#pragma unroll
  for (int mi = 0; mi < 2; ++mi) {
    const int rowb = mBase + r0 + mi * 16 + hi * 8;
    v8f accL = mi ? acc10 : acc00;
    v8f accR = mi ? acc11 : acc01;
#pragma unroll
    for (int v = 0; v < 8; ++v) {
      int row = rowb + v;
      float oL = accL[v] + bvA;
      float oR = accR[v] + bvB;
      if (res) {
        oL += res[(size_t)row * N + colA];
        oR += res[(size_t)row * N + colB];
      }
      C[(size_t)row * N + colA] = oL;
      C[(size_t)row * N + colB] = oR;
    }
  }
}

// ---------------------------------------------------------------------------
// Flash-style sparse-causal spatial attention, register-pipelined staging.
// Q/K/V: [frame*TOK + token, DIMC]. Keys for frame f: frame0 tokens ++ frame
// max(f-1,0) tokens (2048 keys), never materialized. grid(16,4,8), block 128.
// ---------------------------------------------------------------------------
__global__ __launch_bounds__(128) void flash_attn_sparse(
    const float* __restrict__ Q, const float* __restrict__ K,
    const float* __restrict__ V, float* __restrict__ O) {
  __shared__ __align__(16) _Float16 sK[32 * 72];
  __shared__ __align__(16) _Float16 sVt[64 * 40];
  __shared__ __align__(16) _Float16 sP[64 * 40];
  const int t = threadIdx.x;
  const int lane = t & 31, wave = t >> 5;
  const int lh = lane & 15, hi = lane >> 4;
  const int head = blockIdx.y;
  const int fr = blockIdx.z;
  const int qrow = blockIdx.x * 64 + wave * 16 + lh;
  const float* qp = Q + ((size_t)fr * TOK + qrow) * DIMC + head * HDIM;

  v16h aq[2];
#pragma unroll
  for (int i = 0; i < 16; ++i) {
    int d = hi * 16 + i;
    aq[0][i] = (_Float16)(qp[d] * 0.125f);
    aq[1][i] = (_Float16)(qp[32 + d] * 0.125f);
  }
  float m[8], l[8];
  v8f o0 = {}, o1 = {}, o2 = {}, o3 = {};
#pragma unroll
  for (int v = 0; v < 8; ++v) { m[v] = -1e30f; l[v] = 0.f; }

  const int prevf = (fr > 0) ? fr - 1 : 0;

  // ---- preload K-tile 0 into registers ----
  v4f kR[4], vR[4];
#pragma unroll
  for (int i = 0; i < 4; ++i) {
    int e = t + i * 128;                  // 512 quads = 32 keys x 16 d-quads
    int key = e >> 4, dq = (e & 15) * 4;
    int kidx = key;                       // kt = 0 -> frame 0 keys
    size_t g = ((size_t)kidx) * DIMC + head * HDIM + dq;
    kR[i] = *(const v4f*)&K[g];
    vR[i] = *(const v4f*)&V[g];
  }

  for (int kt = 0; kt < 64; ++kt) {
    __syncthreads();   // previous iteration's sVt reads complete
    // ---- commit staged registers to LDS ----
#pragma unroll
    for (int i = 0; i < 4; ++i) {
      int e = t + i * 128;
      int key = e >> 4, dq = (e & 15) * 4;
      *(v4h*)&sK[key * 72 + dq] = cvt4(kR[i]);
#pragma unroll
      for (int j = 0; j < 4; ++j) sVt[(dq + j) * 40 + key] = (_Float16)vR[i][j];
    }
    __syncthreads();

    // ---- issue next tile's loads; they overlap S-WMMA + softmax + PV ----
    if (kt < 63) {
#pragma unroll
      for (int i = 0; i < 4; ++i) {
        int e = t + i * 128;
        int key = e >> 4, dq = (e & 15) * 4;
        int kidx = (kt + 1) * 32 + key;
        int f2 = (kidx < TOK) ? 0 : prevf;
        int tok = kidx & (TOK - 1);
        size_t g = ((size_t)f2 * TOK + tok) * DIMC + head * HDIM + dq;
        kR[i] = *(const v4f*)&K[g];
        vR[i] = *(const v4f*)&V[g];
      }
    }

    v8f s0 = {}, s1 = {};
#pragma unroll
    for (int j = 0; j < 2; ++j) {
      v8h b0lo = *(const v8h*)&sK[lh * 72 + j * 32 + hi * 16];
      v8h b0hi = *(const v8h*)&sK[lh * 72 + j * 32 + hi * 16 + 8];
      s0 = wmma_f16(aq[j], mk16(b0lo, b0hi), s0);
      v8h b1lo = *(const v8h*)&sK[(16 + lh) * 72 + j * 32 + hi * 16];
      v8h b1hi = *(const v8h*)&sK[(16 + lh) * 72 + j * 32 + hi * 16 + 8];
      s1 = wmma_f16(aq[j], mk16(b1lo, b1hi), s1);
    }
#pragma unroll
    for (int v = 0; v < 8; ++v) {
      float mv = fmaxf(s0[v], s1[v]);
#pragma unroll
      for (int msk = 8; msk >= 1; msk >>= 1) mv = fmaxf(mv, __shfl_xor(mv, msk, 32));
      float mn = fmaxf(m[v], mv);
      float p0 = __expf(s0[v] - mn);
      float p1 = __expf(s1[v] - mn);
      float rs = p0 + p1;
#pragma unroll
      for (int msk = 8; msk >= 1; msk >>= 1) rs += __shfl_xor(rs, msk, 32);
      float corr = __expf(m[v] - mn);
      l[v] = l[v] * corr + rs;
      m[v] = mn;
      o0[v] *= corr; o1[v] *= corr; o2[v] *= corr; o3[v] *= corr;
      int prow = wave * 16 + v + hi * 8;
      sP[prow * 40 + lh]      = (_Float16)p0;
      sP[prow * 40 + 16 + lh] = (_Float16)p1;
    }
    __syncthreads();
    v8h plo = *(const v8h*)&sP[(wave * 16 + lh) * 40 + hi * 16];
    v8h phi = *(const v8h*)&sP[(wave * 16 + lh) * 40 + hi * 16 + 8];
    v16h ap = mk16(plo, phi);
    {
      v8h vl, vh;
      vl = *(const v8h*)&sVt[(0 * 16 + lh) * 40 + hi * 16];
      vh = *(const v8h*)&sVt[(0 * 16 + lh) * 40 + hi * 16 + 8];
      o0 = wmma_f16(ap, mk16(vl, vh), o0);
      vl = *(const v8h*)&sVt[(1 * 16 + lh) * 40 + hi * 16];
      vh = *(const v8h*)&sVt[(1 * 16 + lh) * 40 + hi * 16 + 8];
      o1 = wmma_f16(ap, mk16(vl, vh), o1);
      vl = *(const v8h*)&sVt[(2 * 16 + lh) * 40 + hi * 16];
      vh = *(const v8h*)&sVt[(2 * 16 + lh) * 40 + hi * 16 + 8];
      o2 = wmma_f16(ap, mk16(vl, vh), o2);
      vl = *(const v8h*)&sVt[(3 * 16 + lh) * 40 + hi * 16];
      vh = *(const v8h*)&sVt[(3 * 16 + lh) * 40 + hi * 16 + 8];
      o3 = wmma_f16(ap, mk16(vl, vh), o3);
    }
  }
#pragma unroll
  for (int v = 0; v < 8; ++v) {
    int row = blockIdx.x * 64 + wave * 16 + v + hi * 8;
    float inv = 1.f / l[v];
    size_t base = ((size_t)fr * TOK + row) * DIMC + head * HDIM;
    O[base + 0 * 16 + lh] = o0[v] * inv;
    O[base + 1 * 16 + lh] = o1[v] * inv;
    O[base + 2 * 16 + lh] = o2[v] * inv;
    O[base + 3 * 16 + lh] = o3[v] * inv;
  }
}

// ---------------------------------------------------------------------------
// Temporal self-attention over 8 frames (tiny GEMM work: stays VALU), but the
// raw-f32 LDS staging uses CDNA5 async global->LDS copies (ASYNCcnt path):
// no VGPR round-trip, 16B per lane per op. Layout [s][f][c]. grid 1024.
// ---------------------------------------------------------------------------
__global__ __launch_bounds__(256) void temporal_attn(
    const float* __restrict__ Q, const float* __restrict__ K,
    const float* __restrict__ V, float* __restrict__ O) {
  __shared__ __align__(16) float sq[8 * 256];
  __shared__ __align__(16) float sk[8 * 256];
  __shared__ __align__(16) float sv[8 * 256];
  const int s = blockIdx.x;
  const int t = threadIdx.x;

  const unsigned lq = lds_addr32(&sq[0]);
  const unsigned lk = lds_addr32(&sk[0]);
  const unsigned lv = lds_addr32(&sv[0]);
#pragma unroll
  for (int i = 0; i < 2; ++i) {
    int e = t + i * 256;                       // quad index 0..511
    unsigned loff = (unsigned)(e * 16);
    unsigned long long gq = (unsigned long long)(const void*)(Q + (size_t)s * 2048 + e * 4);
    unsigned long long gk = (unsigned long long)(const void*)(K + (size_t)s * 2048 + e * 4);
    unsigned long long gv = (unsigned long long)(const void*)(V + (size_t)s * 2048 + e * 4);
    asm volatile("global_load_async_to_lds_b128 %0, %1, off"
                 :: "v"(lq + loff), "v"(gq) : "memory");
    asm volatile("global_load_async_to_lds_b128 %0, %1, off"
                 :: "v"(lk + loff), "v"(gk) : "memory");
    asm volatile("global_load_async_to_lds_b128 %0, %1, off"
                 :: "v"(lv + loff), "v"(gv) : "memory");
  }
  asm volatile("s_wait_asynccnt 0x0" ::: "memory");
  __syncthreads();

  const int h = t >> 6, dd = t & 63;
  for (int qf = 0; qf < 8; ++qf) {
    float sc[8];
    float mx = -1e30f;
#pragma unroll
    for (int kf = 0; kf < 8; ++kf) {
      float acc = 0.f;
      const float* qp = &sq[qf * 256 + h * 64];
      const float* kp = &sk[kf * 256 + h * 64];
#pragma unroll
      for (int d = 0; d < 64; ++d) acc += qp[d] * kp[d];
      sc[kf] = acc * 0.125f;
      mx = fmaxf(mx, sc[kf]);
    }
    float den = 0.f;
#pragma unroll
    for (int kf = 0; kf < 8; ++kf) { sc[kf] = __expf(sc[kf] - mx); den += sc[kf]; }
    float ov = 0.f;
#pragma unroll
    for (int kf = 0; kf < 8; ++kf) ov += sc[kf] * sv[kf * 256 + h * 64 + dd];
    O[((size_t)s * 8 + qf) * 256 + h * 64 + dd] = ov / den;
  }
}

// ---------------------------------------------------------------------------
// LayerNorm over dim 256 (one row per block).
// ---------------------------------------------------------------------------
__global__ __launch_bounds__(256) void layernorm_k(
    const float* __restrict__ x, const float* __restrict__ g,
    const float* __restrict__ b, float* __restrict__ y) {
  const int row = blockIdx.x, t = threadIdx.x;
  float v = x[(size_t)row * 256 + t];
  float s = v, s2 = v * v;
#pragma unroll
  for (int m = 16; m >= 1; m >>= 1) { s += __shfl_xor(s, m, 32); s2 += __shfl_xor(s2, m, 32); }
  __shared__ float w1[8], w2[8];
  const int lane = t & 31, wv = t >> 5;
  if (lane == 0) { w1[wv] = s; w2[wv] = s2; }
  __syncthreads();
  float S = 0.f, S2 = 0.f;
#pragma unroll
  for (int i = 0; i < 8; ++i) { S += w1[i]; S2 += w2[i]; }
  float mean = S * (1.f / 256.f);
  float var = S2 * (1.f / 256.f) - mean * mean;
  y[(size_t)row * 256 + t] = (v - mean) * rsqrtf(var + 1e-5f) * g[t] + b[t];
}

// ---------------------------------------------------------------------------
// GroupNorm (64 groups of 4 channels x 1024) + SiLU. grid = nImg*64.
// ---------------------------------------------------------------------------
__global__ __launch_bounds__(256) void groupnorm_silu_k(
    const float* __restrict__ x, const float* __restrict__ g,
    const float* __restrict__ b, float* __restrict__ y, int doSilu) {
  const int img = blockIdx.x >> 6;
  const int grp = blockIdx.x & 63;
  const int t = threadIdx.x;
  const size_t base = ((size_t)img * 256 + grp * 4) * 1024;
  float s = 0.f, s2 = 0.f;
  float vals[16];
#pragma unroll
  for (int i = 0; i < 16; ++i) {
    int e = t + i * 256;
    float v = x[base + e];
    vals[i] = v; s += v; s2 += v * v;
  }
#pragma unroll
  for (int m = 16; m >= 1; m >>= 1) { s += __shfl_xor(s, m, 32); s2 += __shfl_xor(s2, m, 32); }
  __shared__ float w1[8], w2[8];
  const int lane = t & 31, wv = t >> 5;
  if (lane == 0) { w1[wv] = s; w2[wv] = s2; }
  __syncthreads();
  float S = 0.f, S2 = 0.f;
#pragma unroll
  for (int i = 0; i < 8; ++i) { S += w1[i]; S2 += w2[i]; }
  float mean = S * (1.f / 4096.f);
  float var = S2 * (1.f / 4096.f) - mean * mean;
  float rstd = rsqrtf(var + 1e-6f);
#pragma unroll
  for (int i = 0; i < 16; ++i) {
    int e = t + i * 256;
    int c = grp * 4 + (e >> 10);
    float v = (vals[i] - mean) * rstd * g[c] + b[c];
    if (doSilu) v = v * sigf(v);
    y[base + e] = v;
  }
}

// ---------------------------------------------------------------------------
// Conv3x3 (Cin=Cout=256, 32x32, pad 1) as implicit GEMM over WMMA.
// Macro-tile 64(pos) x 64(co); 8 waves, per-wave 16x32 (2 WMMAs per step).
// Flattened (tap, ci-chunk) loop, register-pipelined staging.
// grid(4 co-tiles, 16 pos-tiles, nImg). block 256.
// ---------------------------------------------------------------------------
__global__ __launch_bounds__(256) void conv3x3_wmma(
    const float* __restrict__ X, const float* __restrict__ Wt,
    const float* __restrict__ bias, const float* __restrict__ res,
    float* __restrict__ Y) {
  __shared__ __align__(16) _Float16 sA[64 * 40];
  __shared__ __align__(16) _Float16 sB[64 * 40];
  const int t = threadIdx.x;
  const int lane = t & 31, wave = t >> 5;
  const int wm = wave & 3, wn = wave >> 2;
  const int lh = lane & 15, hi = lane >> 4;
  const int img = blockIdx.z;
  const int mBase = blockIdx.y * 64;
  const int nBase = blockIdx.x * 64;
  const int c0 = wn * 32;

  v8f acc0 = {}, acc1 = {};
  float ar[8], br[8];

  // ---- preload iteration 0 (tap 0, ci-chunk 0) ----
  {
    const int dy = -1, dx = -1;
#pragma unroll
    for (int i = 0; i < 8; ++i) {
      int e = t + i * 256;
      int pos = e & 63, ci = e >> 6;
      int p = mBase + pos;
      int py = p >> 5, px = p & 31;
      int sy = py + dy, sx = px + dx;
      float v = 0.f;
      if ((unsigned)sy < 32u && (unsigned)sx < 32u)
        v = X[(((size_t)img * 256) + ci) * 1024 + sy * 32 + sx];
      ar[i] = v;
    }
#pragma unroll
    for (int i = 0; i < 8; ++i) {
      int e = t + i * 256;
      int ci = e & 31, co = e >> 5;
      br[i] = Wt[(((size_t)(nBase + co)) * 256 + ci) * 9];
    }
  }

  for (int it = 0; it < 72; ++it) {        // it = tap*8 + kc
    __syncthreads();
    // ---- commit staged registers to LDS ----
#pragma unroll
    for (int i = 0; i < 8; ++i) {
      int e = t + i * 256;
      int pos = e & 63, ci = e >> 6;
      sA[pos * 40 + ci] = (_Float16)ar[i];
    }
#pragma unroll
    for (int i = 0; i < 8; ++i) {
      int e = t + i * 256;
      int ci = e & 31, co = e >> 5;
      sB[co * 40 + ci] = (_Float16)br[i];
    }
    __syncthreads();

    // ---- issue next iteration's global loads while WMMAs run ----
    if (it + 1 < 72) {
      const int it2 = it + 1;
      const int tap2 = it2 >> 3, kc2 = it2 & 7;
      const int dy = tap2 / 3 - 1, dx = tap2 % 3 - 1;
#pragma unroll
      for (int i = 0; i < 8; ++i) {
        int e = t + i * 256;
        int pos = e & 63, ci = e >> 6;
        int p = mBase + pos;
        int py = p >> 5, px = p & 31;
        int sy = py + dy, sx = px + dx;
        float v = 0.f;
        if ((unsigned)sy < 32u && (unsigned)sx < 32u)
          v = X[(((size_t)img * 256) + kc2 * 32 + ci) * 1024 + sy * 32 + sx];
        ar[i] = v;
      }
#pragma unroll
      for (int i = 0; i < 8; ++i) {
        int e = t + i * 256;
        int ci = e & 31, co = e >> 5;
        br[i] = Wt[(((size_t)(nBase + co)) * 256 + kc2 * 32 + ci) * 9 + tap2];
      }
    }

    v8h alo = *(const v8h*)&sA[(wm * 16 + lh) * 40 + hi * 16];
    v8h ahi = *(const v8h*)&sA[(wm * 16 + lh) * 40 + hi * 16 + 8];
    v8h b0lo = *(const v8h*)&sB[(c0 + lh) * 40 + hi * 16];
    v8h b0hi = *(const v8h*)&sB[(c0 + lh) * 40 + hi * 16 + 8];
    v8h b1lo = *(const v8h*)&sB[(c0 + 16 + lh) * 40 + hi * 16];
    v8h b1hi = *(const v8h*)&sB[(c0 + 16 + lh) * 40 + hi * 16 + 8];
    v16h Af = mk16(alo, ahi);
    acc0 = wmma_f16(Af, mk16(b0lo, b0hi), acc0);
    acc1 = wmma_f16(Af, mk16(b1lo, b1hi), acc1);
  }
  const int p0 = mBase + wm * 16 + hi * 8;
#pragma unroll
  for (int ni = 0; ni < 2; ++ni) {
    const int co = nBase + c0 + ni * 16 + lh;
    const float bv = bias ? bias[co] : 0.f;
    v8f acc = ni ? acc1 : acc0;
#pragma unroll
    for (int v = 0; v < 8; ++v) {
      size_t oidx = (((size_t)img * 256) + co) * 1024 + p0 + v;
      float o = acc[v] + bv;
      if (res) o += res[oidx];
      Y[oidx] = o;
    }
  }
}

// ---------------------------------------------------------------------------
// 1x1 conv 256->1 + sigmoid (gain head). grid(4, nImg).
// ---------------------------------------------------------------------------
__global__ __launch_bounds__(256) void conv1x1_sig(
    const float* __restrict__ X, const float* __restrict__ W,
    const float* __restrict__ b, float* __restrict__ gains) {
  const int img = blockIdx.y;
  const int pos = blockIdx.x * 256 + threadIdx.x;
  float acc = b[0];
#pragma unroll 4
  for (int ci = 0; ci < 256; ++ci)
    acc += W[ci] * X[(((size_t)img * 256) + ci) * 1024 + pos];
  gains[(size_t)img * 1024 + pos] = sigf(acc);
}

// ---------------------------------------------------------------------------
// Elementwise kernels
// ---------------------------------------------------------------------------
__global__ void geglu_k(const float* __restrict__ H, float* __restrict__ G) {
  size_t idx = (size_t)blockIdx.x * 256 + threadIdx.x;  // 8192*1024
  int row = (int)(idx >> 10), j = (int)(idx & 1023);
  float a = H[(size_t)row * 2048 + j];
  float x = H[(size_t)row * 2048 + 1024 + j];
  G[idx] = a * geluf(x);
}

__global__ void blend_k(const float* __restrict__ zt, const float* __restrict__ zp,
                        const float* __restrict__ gain, float* __restrict__ out) {
  int idx = blockIdx.x * 256 + threadIdx.x;  // 262144
  int pos = idx & 1023;
  float k = gain[pos];
  out[idx] = (1.f - k) * zt[idx] + k * sigf(zp[idx]);
}

// Layout shuffles
__global__ void img_to_tok(const float* __restrict__ Z, float* __restrict__ X) {
  int idx = blockIdx.x * 256 + threadIdx.x;  // 2M
  int c = idx & 255, rowt = idx >> 8;
  int img = rowt >> 10, pos = rowt & 1023;
  X[idx] = Z[(((size_t)img << 8) + c) * 1024 + pos];
}
__global__ void tok_to_img(const float* __restrict__ X, float* __restrict__ Y) {
  int idx = blockIdx.x * 256 + threadIdx.x;
  int img = idx >> 18, rem = idx & 262143;
  int c = rem >> 10, pos = rem & 1023;
  Y[idx] = X[(((size_t)img << 10) + pos) * 256 + c];
}
__global__ void perm_fw(const float* __restrict__ X, float* __restrict__ T) {
  int idx = blockIdx.x * 256 + threadIdx.x;
  int c = idx & 255, row = idx >> 8;
  int fr = row & 7, s = row >> 3;
  T[idx] = X[(((size_t)fr << 10) + s) * 256 + c];
}
__global__ void perm_bw(const float* __restrict__ T, float* __restrict__ X) {
  int idx = blockIdx.x * 256 + threadIdx.x;
  int c = idx & 255, row = idx >> 8;
  int fr = row >> 10, s = row & 1023;
  X[idx] = T[(((size_t)s << 3) + fr) * 256 + c];
}

// ---------------------------------------------------------------------------
// Orchestration
// ---------------------------------------------------------------------------
extern "C" void kernel_launch(void* const* d_in, const int* in_sizes, int n_in,
                              void* d_out, int out_size, void* d_ws, size_t ws_size,
                              hipStream_t stream) {
  (void)in_sizes; (void)n_in; (void)out_size; (void)ws_size;
  int cur = 0;
  auto nx = [&]() -> const float* { return (const float*)d_in[cur++]; };

  const float* z = nx();  // z_codes (1,8,256,32,32)

  struct AttnP { const float *wq, *wk, *wv, *wo, *bo; };
  struct BlkP {
    const float *n1g, *n1b; AttnP a1; const float *n3g, *n3b;
    const float *w1, *b1, *w2, *b2; const float *ntg, *ntb; AttnP at;
  };
  BlkP blk[4];
  for (int i = 0; i < 4; ++i) {
    BlkP& B = blk[i];
    B.n1g = nx(); B.n1b = nx();
    B.a1.wq = nx(); B.a1.wk = nx(); B.a1.wv = nx(); B.a1.wo = nx(); B.a1.bo = nx();
    B.n3g = nx(); B.n3b = nx();
    B.w1 = nx(); B.b1 = nx(); B.w2 = nx(); B.b2 = nx();
    B.ntg = nx(); B.ntb = nx();
    B.at.wq = nx(); B.at.wk = nx(); B.at.wv = nx(); B.at.wo = nx(); B.at.bo = nx();
  }
  struct ResP { const float *n1g, *n1b, *c1w, *c1b, *n2g, *n2b, *c2w, *c2b; };
  ResP gres[3];
  for (int i = 0; i < 3; ++i) {
    ResP& R = gres[i];
    R.n1g = nx(); R.n1b = nx(); R.c1w = nx(); R.c1b = nx();
    R.n2g = nx(); R.n2b = nx(); R.c2w = nx(); R.c2b = nx();
  }
  const float* gow = nx(); const float* gob = nx();
  ResP pres[2];
  for (int i = 0; i < 2; ++i) {
    ResP& R = pres[i];
    R.n1g = nx(); R.n1b = nx(); R.c1w = nx(); R.c1b = nx();
    R.n2g = nx(); R.n2b = nx(); R.c2w = nx(); R.c2b = nx();
  }

  float* Wp = (float*)d_ws;
  size_t off = 0;
  auto alloc = [&](size_t n) { float* p = Wp + off; off += n; return p; };
  const size_t TKN = (size_t)NROWS * DIMC;
  float* X  = alloc(TKN);
  float* NB = alloc(TKN);
  float* Qb = alloc(TKN);
  float* Kb = alloc(TKN);
  float* Vb = alloc(TKN);
  float* Ob = alloc(TKN);
  float* Tb = alloc(TKN);
  float* Hb = alloc((size_t)NROWS * FFH);
  float* Gb = alloc((size_t)NROWS * FFO);
  float* IA = alloc(TKN);
  float* IB = alloc(TKN);
  float* IC = alloc(TKN);
  float* GN = alloc((size_t)N_FRAMES * 1024);
  float* PA = alloc((size_t)256 * 1024);
  float* PB = alloc((size_t)256 * 1024);
  float* PC = alloc((size_t)256 * 1024);

  img_to_tok<<<8192, 256, 0, stream>>>(z, X);

  for (int L = 0; L < 4; ++L) {
    const BlkP& B = blk[L];
    // spatial sparse-causal attention
    layernorm_k<<<8192, 256, 0, stream>>>(X, B.n1g, B.n1b, NB);
    gemm_wmma<0><<<dim3(4, 64), 256, 0, stream>>>(NB, B.a1.wq, nullptr, nullptr, Qb, NROWS, 256, 256);
    gemm_wmma<0><<<dim3(4, 64), 256, 0, stream>>>(NB, B.a1.wk, nullptr, nullptr, Kb, NROWS, 256, 256);
    gemm_wmma<0><<<dim3(4, 64), 256, 0, stream>>>(NB, B.a1.wv, nullptr, nullptr, Vb, NROWS, 256, 256);
    flash_attn_sparse<<<dim3(16, 4, 8), 128, 0, stream>>>(Qb, Kb, Vb, Ob);
    gemm_wmma<0><<<dim3(4, 64), 256, 0, stream>>>(Ob, B.a1.wo, B.a1.bo, X, X, NROWS, 256, 256);
    // GEGLU FF
    layernorm_k<<<8192, 256, 0, stream>>>(X, B.n3g, B.n3b, NB);
    gemm_wmma<0><<<dim3(32, 64), 256, 0, stream>>>(NB, B.w1, B.b1, nullptr, Hb, NROWS, FFH, 256);
    geglu_k<<<32768, 256, 0, stream>>>(Hb, Gb);
    gemm_wmma<0><<<dim3(4, 64), 256, 0, stream>>>(Gb, B.w2, B.b2, X, X, NROWS, 256, FFO);
    // temporal attention
    perm_fw<<<8192, 256, 0, stream>>>(X, Tb);
    layernorm_k<<<8192, 256, 0, stream>>>(Tb, B.ntg, B.ntb, NB);
    gemm_wmma<0><<<dim3(4, 64), 256, 0, stream>>>(NB, B.at.wq, nullptr, nullptr, Qb, NROWS, 256, 256);
    gemm_wmma<0><<<dim3(4, 64), 256, 0, stream>>>(NB, B.at.wk, nullptr, nullptr, Kb, NROWS, 256, 256);
    gemm_wmma<0><<<dim3(4, 64), 256, 0, stream>>>(NB, B.at.wv, nullptr, nullptr, Vb, NROWS, 256, 256);
    temporal_attn<<<1024, 256, 0, stream>>>(Qb, Kb, Vb, Ob);
    gemm_wmma<0><<<dim3(4, 64), 256, 0, stream>>>(Ob, B.at.wo, B.at.bo, Tb, Tb, NROWS, 256, 256);
    perm_bw<<<8192, 256, 0, stream>>>(Tb, X);
  }

  tok_to_img<<<8192, 256, 0, stream>>>(X, IA);

  for (int rI = 0; rI < 3; ++rI) {
    const ResP& R = gres[rI];
    groupnorm_silu_k<<<8 * 64, 256, 0, stream>>>(IA, R.n1g, R.n1b, IB, 1);
    conv3x3_wmma<<<dim3(4, 16, 8), 256, 0, stream>>>(IB, R.c1w, R.c1b, nullptr, IC);
    groupnorm_silu_k<<<8 * 64, 256, 0, stream>>>(IC, R.n2g, R.n2b, IB, 1);
    conv3x3_wmma<<<dim3(4, 16, 8), 256, 0, stream>>>(IB, R.c2w, R.c2b, IA, IA);
  }
  conv1x1_sig<<<dim3(4, 8), 256, 0, stream>>>(IA, gow, gob, GN);

  float* out = (float*)d_out;
  const size_t FSZ = (size_t)256 * 1024;
  hipMemcpyAsync(out, z, FSZ * sizeof(float), hipMemcpyDeviceToDevice, stream);

  for (int ts = 1; ts < 8; ++ts) {
    const float* zhat = out + (size_t)(ts - 1) * FSZ;
    {
      const ResP& R = pres[0];
      groupnorm_silu_k<<<64, 256, 0, stream>>>(zhat, R.n1g, R.n1b, PA, 1);
      conv3x3_wmma<<<dim3(4, 16, 1), 256, 0, stream>>>(PA, R.c1w, R.c1b, nullptr, PB);
      groupnorm_silu_k<<<64, 256, 0, stream>>>(PB, R.n2g, R.n2b, PA, 1);
      conv3x3_wmma<<<dim3(4, 16, 1), 256, 0, stream>>>(PA, R.c2w, R.c2b, zhat, PB);
    }
    {
      const ResP& R = pres[1];
      groupnorm_silu_k<<<64, 256, 0, stream>>>(PB, R.n1g, R.n1b, PA, 1);
      conv3x3_wmma<<<dim3(4, 16, 1), 256, 0, stream>>>(PA, R.c1w, R.c1b, nullptr, PC);
      groupnorm_silu_k<<<64, 256, 0, stream>>>(PC, R.n2g, R.n2b, PA, 1);
      conv3x3_wmma<<<dim3(4, 16, 1), 256, 0, stream>>>(PA, R.c2w, R.c2b, PB, PC);
    }
    blend_k<<<1024, 256, 0, stream>>>(z + (size_t)ts * FSZ, PC, GN + (size_t)ts * 1024,
                                      out + (size_t)ts * FSZ);
  }
}
